// LeanSelfAttention_73942156967951
// MI455X (gfx1250) — matrix operand order repaired and checked by reading, not compiled, not run on hardware
//
#include <hip/hip_runtime.h>
#include <hip/hip_bf16.h>

// B=4, S=2048, H=1024, NH=16, HD=64
#define Bq 4
#define Sq 2048
#define Hq 1024
#define NHq 16
#define HDq 64
#define ROWS (Bq * Sq)          // 8192
#define H3 (3 * Hq)             // 3072

typedef __bf16 bf16;
typedef __attribute__((ext_vector_type(16))) __bf16 v16bf;
typedef __attribute__((ext_vector_type(8)))  __bf16 bf16x8;
typedef __attribute__((ext_vector_type(8)))  float  v8f;

__device__ __forceinline__ v8f wmma_bf16(v16bf a, v16bf b, v8f c) {
  // D(16x16 f32) = A(16x32 bf16) * B(32x16 bf16) + C
  return __builtin_amdgcn_wmma_f32_16x16x32_bf16(
      /*neg_a=*/false, a, /*neg_b=*/false, b,
      /*c_mod=*/(short)0, c, /*reuse_a=*/false, /*reuse_b=*/false);
}

// A fragment: 16x32 bf16 tile at (m0,k0) from row-major [*, ld] bf16.
// lane l (<16): row M=l, K = {k0+0..7, k0+16..23}; l>=16: K += 8.
__device__ __forceinline__ v16bf load_afrag(const bf16* base, int ld,
                                            int m0, int k0, int lane) {
  int m = m0 + (lane & 15);
  int koff = (lane >= 16) ? 8 : 0;
  const bf16* p = base + (size_t)m * ld + k0 + koff;
  bf16x8 lo = *(const bf16x8*)(p);
  bf16x8 hi = *(const bf16x8*)(p + 16);
  v16bf a;
#pragma unroll
  for (int i = 0; i < 8; ++i) { a[i] = lo[i]; a[i + 8] = hi[i]; }
  return a;
}

// B fragment: 32x16 bf16 tile, element (kk, n). Source stored TRANSPOSED:
// baseT[n * ld + kk]. lane l: N = n0 + l%16, K = {k0+0..7, k0+16..23} + 8*(l>=16).
__device__ __forceinline__ v16bf load_bfragT(const bf16* baseT, int ld,
                                             int n0, int k0, int lane) {
  int n = n0 + (lane & 15);
  int kk = k0 + ((lane >= 16) ? 8 : 0);
  const bf16* p = baseT + (size_t)n * ld + kk;
  bf16x8 lo = *(const bf16x8*)(p);
  bf16x8 hi = *(const bf16x8*)(p + 16);
  v16bf b;
#pragma unroll
  for (int i = 0; i < 8; ++i) { b[i] = lo[i]; b[i + 8] = hi[i]; }
  return b;
}

// ---------------- Kernel 1: LayerNorm -> bf16 ----------------
__global__ __launch_bounds__(256) void ln_kernel(const float* __restrict__ x,
                                                 const float* __restrict__ gamma,
                                                 const float* __restrict__ beta,
                                                 bf16* __restrict__ xln) {
  __shared__ float red[8];
  int row = blockIdx.x;  // [0, ROWS)
  int tid = threadIdx.x;
  const float* xr = x + (size_t)row * Hq;
  float4 v = ((const float4*)xr)[tid];  // 256 threads * 4 = 1024
  float s = v.x + v.y + v.z + v.w;
#pragma unroll
  for (int off = 16; off > 0; off >>= 1) s += __shfl_down(s, off, 32);
  if ((tid & 31) == 0) red[tid >> 5] = s;
  __syncthreads();
  if (tid == 0) {
    float t = 0.f;
#pragma unroll
    for (int i = 0; i < 8; ++i) t += red[i];
    red[0] = t * (1.0f / Hq);
  }
  __syncthreads();
  float mu = red[0];
  __syncthreads();
  float dx = v.x - mu, dy = v.y - mu, dz = v.z - mu, dw = v.w - mu;
  float s2 = dx * dx + dy * dy + dz * dz + dw * dw;
#pragma unroll
  for (int off = 16; off > 0; off >>= 1) s2 += __shfl_down(s2, off, 32);
  if ((tid & 31) == 0) red[tid >> 5] = s2;
  __syncthreads();
  if (tid == 0) {
    float t = 0.f;
#pragma unroll
    for (int i = 0; i < 8; ++i) t += red[i];
    red[0] = rsqrtf(t * (1.0f / Hq) + 1e-12f);
  }
  __syncthreads();
  float rinv = red[0];
  int c = tid * 4;
  bf16* o = xln + (size_t)row * Hq + c;
  o[0] = (bf16)(dx * rinv * gamma[c + 0] + beta[c + 0]);
  o[1] = (bf16)(dy * rinv * gamma[c + 1] + beta[c + 1]);
  o[2] = (bf16)(dz * rinv * gamma[c + 2] + beta[c + 2]);
  o[3] = (bf16)(dw * rinv * gamma[c + 3] + beta[c + 3]);
}

// ---------- Kernel 2: transpose + f32->bf16 weights: out[n][k] = in[k][n] ----------
__global__ __launch_bounds__(1024) void transpose_bf16(const float* __restrict__ in,
                                                       bf16* __restrict__ out,
                                                       int K, int N) {
  __shared__ float tile[32][33];
  int n0 = blockIdx.x * 32, k0 = blockIdx.y * 32;
  int tx = threadIdx.x, ty = threadIdx.y;
  tile[ty][tx] = in[(size_t)(k0 + ty) * N + n0 + tx];
  __syncthreads();
  out[(size_t)(n0 + ty) * K + k0 + tx] = (bf16)tile[tx][ty];
}

// ---------------- Shared GEMM helpers (32x64 wave tile) ----------------
__device__ __forceinline__ void load_frags(const bf16* __restrict__ A,
                                           const bf16* __restrict__ BT,
                                           int m0, int n0, int k, int lane,
                                           v16bf& a0, v16bf& a1, v16bf b[4]) {
  a0 = load_afrag(A, Hq, m0,      k, lane);
  a1 = load_afrag(A, Hq, m0 + 16, k, lane);
#pragma unroll
  for (int t = 0; t < 4; ++t) b[t] = load_bfragT(BT, Hq, n0 + t * 16, k, lane);
}

__device__ __forceinline__ void gemm_step(v8f acc[2][4], const v16bf& a0,
                                          const v16bf& a1, const v16bf b[4]) {
#pragma unroll
  for (int t = 0; t < 4; ++t) {
    acc[0][t] = wmma_bf16(a0, b[t], acc[0][t]);
    acc[1][t] = wmma_bf16(a1, b[t], acc[1][t]);
  }
}

// Ping-pong pipelined K loop: no register copies, loads of step k+32 overlap
// the 8 WMMAs of step k.
__device__ __forceinline__ void gemm_mainloop(const bf16* __restrict__ A,
                                              const bf16* __restrict__ BT,
                                              int m0, int n0, int lane,
                                              v8f acc[2][4]) {
  v16bf a0s0, a1s0, bs0[4];
  v16bf a0s1, a1s1, bs1[4];
  load_frags(A, BT, m0, n0, 0, lane, a0s0, a1s0, bs0);
  for (int k = 0; k < Hq - 64; k += 64) {
    int kp = (k + 192 < Hq) ? k + 192 : k;   // L2 prefetch ahead
    __builtin_prefetch((const void*)(A  + (size_t)(m0 + (lane & 15)) * Hq + kp), 0, 1);
    __builtin_prefetch((const void*)(BT + (size_t)(n0 + (lane & 15)) * Hq + kp), 0, 1);
    load_frags(A, BT, m0, n0, k + 32, lane, a0s1, a1s1, bs1);
    gemm_step(acc, a0s0, a1s0, bs0);
    load_frags(A, BT, m0, n0, k + 64, lane, a0s0, a1s0, bs0);
    gemm_step(acc, a0s1, a1s1, bs1);
  }
  // peeled last two k-steps: slot0 holds Hq-64
  load_frags(A, BT, m0, n0, Hq - 32, lane, a0s1, a1s1, bs1);
  gemm_step(acc, a0s0, a1s0, bs0);
  gemm_step(acc, a0s1, a1s1, bs1);
}

// ---------------- Kernel 3: QKV GEMM (WMMA) + head scatter ----------------
// C[8192 x 3072] = xln[8192x1024] @ Wqkv.
// Q,K stored [B][NH][S][HD]; V stored transposed [B][NH][HD][S] (all bf16).
__global__ __launch_bounds__(256) void gemm_qkv(const bf16* __restrict__ A,
                                                const bf16* __restrict__ BT,
                                                const float* __restrict__ bias,
                                                bf16* __restrict__ Qh,
                                                bf16* __restrict__ Kh,
                                                bf16* __restrict__ Vt) {
  int lane = threadIdx.x & 31, wave = threadIdx.x >> 5;
  int gw = blockIdx.x * 8 + wave;
  const int NT = H3 / 64;  // 48
  int mt = gw / NT, nt = gw % NT;
  int m0 = mt * 32, n0 = nt * 64;
  v8f acc[2][4] = {};
  gemm_mainloop(A, BT, m0, n0, lane, acc);
  int half = lane >> 4, nl = lane & 15;
#pragma unroll
  for (int mi = 0; mi < 2; ++mi) {
#pragma unroll
    for (int t = 0; t < 4; ++t) {
      int n = n0 + t * 16 + nl;
      float bb = bias[n];
      int part = n >> 10;      // which of q/k/v
      int rem  = n & 1023;
      int head = rem >> 6;
      int d    = rem & 63;
#pragma unroll
      for (int r = 0; r < 8; ++r) {
        int m = m0 + mi * 16 + r + half * 8;   // global row in [0, 8192)
        int b_ = m >> 11;                      // /S
        int s_ = m & (Sq - 1);
        bf16 bv = (bf16)(acc[mi][t][r] + bb);
        size_t bh = (size_t)b_ * NHq + head;
        if (part == 0)      Qh[(bh * Sq + s_) * HDq + d] = bv;
        else if (part == 1) Kh[(bh * Sq + s_) * HDq + d] = bv;
        else                Vt[(bh * HDq + d) * Sq + s_] = bv;
      }
    }
  }
}

// ---------------- Kernel 4: causal FlashAttention (WMMA) ----------------
// block = 256 threads = 8 waves = 128 query rows of one (b,h).
// Key blocks of 32 are ping-pong double-buffered: block j+1's K/V fragments
// load while block j computes (scores -> softmax -> LDS P re-layout -> PV).
__device__ __forceinline__ void attn_load_kv(const bf16* __restrict__ K,
                                             const bf16* __restrict__ V,
                                             int kc, int lane,
                                             v16bf& k0, v16bf& k1,
                                             v16bf& k2, v16bf& k3, v16bf v[4]) {
  k0 = load_bfragT(K, HDq, kc,      0,  lane);
  k1 = load_bfragT(K, HDq, kc,      32, lane);
  k2 = load_bfragT(K, HDq, kc + 16, 0,  lane);
  k3 = load_bfragT(K, HDq, kc + 16, 32, lane);
#pragma unroll
  for (int t = 0; t < 4; ++t) v[t] = load_bfragT(V, Sq, t * 16, kc, lane);
}

__device__ __forceinline__ void attn_process(int kc, int q0, int half, int nl,
                                             bf16* myP,
                                             const v16bf& aQ0, const v16bf& aQ1,
                                             const v16bf& bK0, const v16bf& bK1,
                                             const v16bf& bK2, const v16bf& bK3,
                                             const v16bf bV[4],
                                             v8f acc[4], float mstat[8],
                                             float lstat[8]) {
  // S = Q K^T : two 16x16 column tiles, two k-steps over HD=64
  v8f c0 = {}, c1 = {};
  c0 = wmma_bf16(aQ0, bK0, c0);
  c0 = wmma_bf16(aQ1, bK1, c0);
  c1 = wmma_bf16(aQ0, bK2, c1);
  c1 = wmma_bf16(aQ1, bK3, c1);
  int col0 = kc + nl, col1 = kc + 16 + nl;
#pragma unroll
  for (int r = 0; r < 8; ++r) {
    int row = q0 + r + half * 8;
    float s0 = c0[r] * 0.125f; if (col0 > row) s0 = -10000.f;
    float s1 = c1[r] * 0.125f; if (col1 > row) s1 = -10000.f;
    float mx = fmaxf(s0, s1);
#pragma unroll
    for (int off = 1; off < 16; off <<= 1) mx = fmaxf(mx, __shfl_xor(mx, off, 16));
    float mn = fmaxf(mstat[r], mx);
    float alpha = __expf(mstat[r] - mn);
    float p0 = __expf(s0 - mn), p1 = __expf(s1 - mn);
    float rs = p0 + p1;
#pragma unroll
    for (int off = 1; off < 16; off <<= 1) rs += __shfl_xor(rs, off, 16);
    lstat[r] = lstat[r] * alpha + rs;
    mstat[r] = mn;
#pragma unroll
    for (int t = 0; t < 4; ++t) acc[t][r] *= alpha;
    int pr = r + half * 8;  // C-layout row this lane-half owns
    myP[pr * 32 + nl]      = (bf16)p0;
    myP[pr * 32 + 16 + nl] = (bf16)p1;
  }
  __syncthreads();
  // Re-read P from LDS in A-fragment layout (16x32, K = key position)
  v16bf aP;
  {
    const bf16* p = myP + nl * 32 + (half ? 8 : 0);
    bf16x8 lo = *(const bf16x8*)(p);
    bf16x8 hi = *(const bf16x8*)(p + 16);
#pragma unroll
    for (int i = 0; i < 8; ++i) { aP[i] = lo[i]; aP[i + 8] = hi[i]; }
  }
  __syncthreads();
  // O += P @ V : V^T is [HD][S]; bV loaded a full block ago, resident
#pragma unroll
  for (int t = 0; t < 4; ++t) acc[t] = wmma_bf16(aP, bV[t], acc[t]);
}

__global__ __launch_bounds__(256) void attn_kernel(const bf16* __restrict__ Qh,
                                                   const bf16* __restrict__ Kh,
                                                   const bf16* __restrict__ Vt,
                                                   bf16* __restrict__ attn) {
  __shared__ __align__(16) bf16 ldsP[8][16 * 32];  // per-wave P tile (16x32)
  int lane = threadIdx.x & 31, wave = threadIdx.x >> 5;
  const int QB = Sq / 128;  // 16 query blocks per (b,h)
  int qblk = blockIdx.x % QB;
  int bh   = blockIdx.x / QB;
  int h = bh & (NHq - 1), b = bh >> 4;
  const bf16* Q = Qh + (size_t)bh * Sq * HDq;
  const bf16* K = Kh + (size_t)bh * Sq * HDq;
  const bf16* V = Vt + (size_t)bh * HDq * Sq;  // [HD][S]
  int q0 = qblk * 128 + wave * 16;
  v16bf aQ0 = load_afrag(Q, HDq, q0, 0, lane);
  v16bf aQ1 = load_afrag(Q, HDq, q0, 32, lane);
  v8f acc[4] = {};
  float mstat[8], lstat[8];
#pragma unroll
  for (int r = 0; r < 8; ++r) { mstat[r] = -1e30f; lstat[r] = 0.f; }
  int half = lane >> 4, nl = lane & 15;
  bf16* myP = &ldsP[wave][0];
  int jmax = (qblk + 1) * 4;  // uniform over block; multiple of 4
  // ping-pong KV fragment buffers
  v16bf ka0, ka1, ka2, ka3, va[4];
  v16bf kb0, kb1, kb2, kb3, vb[4];
  attn_load_kv(K, V, 0, lane, ka0, ka1, ka2, ka3, va);
  int j = 0;
  for (; j < jmax - 2; j += 2) {
    attn_load_kv(K, V, (j + 1) * 32, lane, kb0, kb1, kb2, kb3, vb);
    attn_process(j * 32, q0, half, nl, myP, aQ0, aQ1,
                 ka0, ka1, ka2, ka3, va, acc, mstat, lstat);
    attn_load_kv(K, V, (j + 2) * 32, lane, ka0, ka1, ka2, ka3, va);
    attn_process((j + 1) * 32, q0, half, nl, myP, aQ0, aQ1,
                 kb0, kb1, kb2, kb3, vb, acc, mstat, lstat);
  }
  // peeled final two key blocks (slot A holds block jmax-2)
  attn_load_kv(K, V, (jmax - 1) * 32, lane, kb0, kb1, kb2, kb3, vb);
  attn_process((jmax - 2) * 32, q0, half, nl, myP, aQ0, aQ1,
               ka0, ka1, ka2, ka3, va, acc, mstat, lstat);
  attn_process((jmax - 1) * 32, q0, half, nl, myP, aQ0, aQ1,
               kb0, kb1, kb2, kb3, vb, acc, mstat, lstat);
  // normalize and store to attn [B*S][H] bf16 (A operand of out-proj GEMM)
#pragma unroll
  for (int r = 0; r < 8; ++r) {
    float inv = 1.0f / lstat[r];
    int row = q0 + r + half * 8;
    size_t orow = ((size_t)b * Sq + row) * Hq + h * HDq;
#pragma unroll
    for (int t = 0; t < 4; ++t)
      attn[orow + t * 16 + nl] = (bf16)(acc[t][r] * inv);
  }
}

// -------- Kernel 5: output projection GEMM + bias + residual (f32 out) --------
__global__ __launch_bounds__(256) void gemm_out(const bf16* __restrict__ A,
                                                const bf16* __restrict__ BT,
                                                const float* __restrict__ bias,
                                                const float* __restrict__ resid,
                                                float* __restrict__ out) {
  int lane = threadIdx.x & 31, wave = threadIdx.x >> 5;
  int gw = blockIdx.x * 8 + wave;
  const int NT = Hq / 64;  // 16
  int mt = gw / NT, nt = gw % NT;
  int m0 = mt * 32, n0 = nt * 64;
  v8f acc[2][4] = {};
  gemm_mainloop(A, BT, m0, n0, lane, acc);
  int half = lane >> 4, nl = lane & 15;
#pragma unroll
  for (int mi = 0; mi < 2; ++mi) {
#pragma unroll
    for (int t = 0; t < 4; ++t) {
      int n = n0 + t * 16 + nl;
      float bb = bias[n];
#pragma unroll
      for (int r = 0; r < 8; ++r) {
        int m = m0 + mi * 16 + r + half * 8;
        size_t idx = (size_t)m * Hq + n;
        out[idx] = acc[mi][t][r] + bb + resid[idx];
      }
    }
  }
}

extern "C" void kernel_launch(void* const* d_in, const int* in_sizes, int n_in,
                              void* d_out, int out_size, void* d_ws, size_t ws_size,
                              hipStream_t stream) {
  const float* hidden   = (const float*)d_in[0];
  const float* ln_scale = (const float*)d_in[1];
  const float* ln_bias  = (const float*)d_in[2];
  const float* W_qkv    = (const float*)d_in[3];
  const float* b_qkv    = (const float*)d_in[4];
  const float* W_out    = (const float*)d_in[5];
  const float* b_out    = (const float*)d_in[6];
  float* out = (float*)d_out;

  // Workspace layout (bytes); total = 88 MiB
  char* ws = (char*)d_ws;
  const size_t sz_xln   = (size_t)ROWS * Hq * 2;          // 16 MiB
  const size_t sz_WqkvT = (size_t)H3 * Hq * 2;            //  6 MiB
  const size_t sz_WoutT = (size_t)Hq * Hq * 2;            //  2 MiB
  const size_t sz_head  = (size_t)Bq * NHq * Sq * HDq * 2;// 16 MiB each
  bf16* xln   = (bf16*)(ws);                 ws += sz_xln;
  bf16* WqkvT = (bf16*)(ws);                 ws += sz_WqkvT;
  bf16* WoutT = (bf16*)(ws);                 ws += sz_WoutT;
  bf16* Qh    = (bf16*)(ws);                 ws += sz_head;
  bf16* Kh    = (bf16*)(ws);                 ws += sz_head;
  bf16* Vt    = (bf16*)(ws);                 ws += sz_head;
  bf16* attn  = (bf16*)(ws);                 ws += sz_xln;

  ln_kernel<<<ROWS, 256, 0, stream>>>(hidden, ln_scale, ln_bias, xln);
  transpose_bf16<<<dim3(H3 / 32, Hq / 32), dim3(32, 32), 0, stream>>>(W_qkv, WqkvT, Hq, H3);
  transpose_bf16<<<dim3(Hq / 32, Hq / 32), dim3(32, 32), 0, stream>>>(W_out, WoutT, Hq, Hq);
  // 256 M-tiles * 48 N-strips = 12288 waves / 8 per block
  gemm_qkv<<<1536, 256, 0, stream>>>(xln, WqkvT, b_qkv, Qh, Kh, Vt);
  // B*NH*(S/128) = 1024 blocks
  attn_kernel<<<1024, 256, 0, stream>>>(Qh, Kh, Vt, attn);
  // 256 M-tiles * 16 N-strips = 4096 waves / 8 per block
  gemm_out<<<512, 256, 0, stream>>>(attn, WoutT, b_out, hidden, out);
}
